// CrossCRFLoss_18915035972252
// MI455X (gfx1250) — matrix operand order repaired and checked by reading, not compiled, not run on hardware
//
#include <hip/hip_runtime.h>
#include <hip/hip_bf16.h>

typedef __attribute__((ext_vector_type(16))) _Float16 v16h;
typedef __attribute__((ext_vector_type(8)))  _Float16 v8h;
typedef __attribute__((ext_vector_type(8)))  float    v8f;

#define NV 128
#define TT 256
#define LL 512
#define SS 8
#define ROWS 16          // batch rows per workgroup
#define NWG  (NV / ROWS) // 8 workgroups
#define NTHR 512         // 16 waves of 32
#define NEG_INF_F (-10000.0f)

// dynamic LDS layout (bytes)
#define LDS_ALPHA_OFF 0
#define LDS_AEXP_OFF  (ROWS * LL * 4)                       // 32768
#define LDS_PENAL_OFF (LDS_AEXP_OFF + ROWS * LL * 2)        // 49152
#define LDS_TOTAL     (LDS_PENAL_OFF + ROWS * LL * 4)       // 81920

union AFrag { v16h v; v8h h[2]; };

// ---------------------------------------------------------------------------
// Kernel 1: semlink penalty mask -> penal[v][j] in {0, NEG_INF}
// ---------------------------------------------------------------------------
__global__ __launch_bounds__(LL)
void semlink_kernel(const int* __restrict__ semlink,   // [NV][2][SS]
                    const int* __restrict__ srl_b2i,   // [100]
                    const int* __restrict__ vn_b2i,    // [200]
                    const int* __restrict__ srl2c,     // [100][LL] bool-as-int
                    const int* __restrict__ vn2c,      // [200][LL]
                    const int* __restrict__ content,   // [LL]
                    float* __restrict__ penal)         // [NV][LL]
{
    const int v = blockIdx.x;
    const int j = threadIdx.x;
    int roles[2 * SS], args[2 * SS];
    for (int s = 0; s < SS; ++s) {
        int braw = semlink[(v * 2 + 0) * SS + s];
        int b_role = (braw != -1) ? braw : 0;
        int araw = semlink[(v * 2 + 1) * SS + s];
        int b_arg = (araw != -1) ? araw : 0;
        roles[s]      = b_role;
        roles[SS + s] = srl_b2i[b_role];
        args[s]       = b_arg;
        args[SS + s]  = vn_b2i[b_arg];
    }
    bool valid = false;
    for (int k = 0; k < 2 * SS; ++k) valid = valid || (roles[k] != 0);
    bool inner = false;
    for (int k = 0; k < 2 * SS; ++k)
        inner = inner || ((srl2c[roles[k] * LL + j] != 0) && (vn2c[args[k] * LL + j] != 0));
    bool disable = (content[j] != 0) && !inner && valid;
    penal[v * LL + j] = disable ? NEG_INF_F : 0.0f;
}

// ---------------------------------------------------------------------------
// Kernel 2: Bt[j][i] = (f16) exp(transitions[i][j])   (transposed for WMMA-B)
// ---------------------------------------------------------------------------
__global__ __launch_bounds__(LL)
void expT_kernel(const float* __restrict__ transitions, _Float16* __restrict__ Bt)
{
    const int j = blockIdx.x;       // Bt row
    const int i = threadIdx.x;      // Bt col
    Bt[j * LL + i] = (_Float16)__expf(transitions[i * LL + j]);
}

// ---------------------------------------------------------------------------
// Kernel 3: the CRF forward scan as 255 batched WMMA GEMMs + gold score tail.
// One workgroup per 16 batch rows; 16 waves; each wave owns 2 N-tiles of 16.
// ---------------------------------------------------------------------------
__global__ __launch_bounds__(NTHR)
void crf_scan_kernel(const float* __restrict__ label_score,   // [NV][TT][LL]
                     const int*   __restrict__ tags,          // [NV][TT]
                     const int*   __restrict__ mask,          // [NV][TT]
                     const float* __restrict__ transitions,   // [LL][LL]
                     const float* __restrict__ start_t,       // [LL]
                     const float* __restrict__ end_t,         // [LL]
                     const float* __restrict__ penal,         // [NV][LL]
                     const _Float16* __restrict__ Bt,         // [LL][LL] = exp(T)^T
                     float* __restrict__ partials)            // [NWG]
{
    extern __shared__ char smem[];
    float*    alpha   = (float*)(smem + LDS_ALPHA_OFF);   // [ROWS][LL] f32
    _Float16* aexp    = (_Float16*)(smem + LDS_AEXP_OFF); // [ROWS][LL] f16
    float*    penal_s = (float*)(smem + LDS_PENAL_OFF);   // [ROWS][LL] f32

    __shared__ float rowmax_s[ROWS];
    __shared__ float maskrow[ROWS];
    __shared__ float logz_s[ROWS];
    __shared__ float contrib[ROWS];

    const int tid  = threadIdx.x;
    const int lane = tid & 31;
    const int wid  = tid >> 5;               // 0..15
    const int v0   = blockIdx.x * ROWS;

    // --- init: stage penal slice in LDS; alpha0 = start_t + score[:,0] + penal ---
    {
        const int row = wid;
        const int b = v0 + row;
        for (int q = lane; q < LL; q += 32) {
            const float p = penal[b * LL + q];
            penal_s[row * LL + q] = p;
            alpha[row * LL + q] = start_t[q] + label_score[(size_t)b * TT * LL + q] + p;
        }
    }
    __syncthreads();

    // WMMA lane mapping (contiguous k: lane half h holds k 16h..16h+15 of A and B)
    const int mrow   = lane & 15;                 // A: row index
    const int koffA  = (lane < 16) ? 0 : 16;      // A: K base of lane half
    const int nlocal = lane & 15;                 // B/C: column index
    const int koffB  = (lane < 16) ? 0 : 16;      // B: K base of lane half
    const int crow   = (lane < 16) ? 0 : 8;       // C: row offset of lane half
    const int nt0    = wid * 2;                   // this wave's first N-tile

    for (int t = 1; t < TT; ++t) {
        // ---- phase A: rowmax + exp(alpha - rowmax) -> f16, per-row mask,
        //               prefetch next step's emission slice ----
        {
            const int row = wid;
            const int b = v0 + row;
            if (t + 1 < TT) {
                // one 64B line per lane covers the 2KB emission row for t+1
                const char* nxt = (const char*)&label_score[((size_t)b * TT + (t + 1)) * LL];
                __builtin_prefetch(nxt + lane * 64, 0, 3);
            }
            float m = -3.4e38f;
            for (int q = lane; q < LL; q += 32) m = fmaxf(m, alpha[row * LL + q]);
            for (int off = 16; off >= 1; off >>= 1)
                m = fmaxf(m, __shfl_xor(m, off));
            for (int q = lane; q < LL; q += 32)
                aexp[row * LL + q] = (_Float16)__expf(alpha[row * LL + q] - m);
            if (lane == 0) {
                rowmax_s[row] = m;
                maskrow[row]  = (mask[b * TT + t] != 0) ? 1.0f : 0.0f;
            }
        }
        __syncthreads();

        // ---- phase B: C = aexp · E via v_wmma_f32_16x16x32_f16 ----
        v8f acc0 = {}; v8f acc1 = {};
        #pragma unroll 4
        for (int kt = 0; kt < LL / 32; ++kt) {
            const int kb = kt * 32;
            AFrag a;
            a.h[0] = *(const v8h*)&aexp[mrow * LL + kb + koffA];
            a.h[1] = *(const v8h*)&aexp[mrow * LL + kb + koffA + 8];
            v16h b0 = *(const v16h*)&Bt[(size_t)((nt0 + 0) * 16 + nlocal) * LL + kb + koffB];
            v16h b1 = *(const v16h*)&Bt[(size_t)((nt0 + 1) * 16 + nlocal) * LL + kb + koffB];
            acc0 = __builtin_amdgcn_wmma_f32_16x16x32_f16(
                       false, a.v, false, b0, (short)0, acc0, false, false);
            acc1 = __builtin_amdgcn_wmma_f32_16x16x32_f16(
                       false, a.v, false, b1, (short)0, acc1, false, false);
        }

        // ---- alpha' = rowmax + log(C) + emit  (masked rows keep old alpha) ----
        #pragma unroll
        for (int tile = 0; tile < 2; ++tile) {
            const v8f acc = tile ? acc1 : acc0;
            const int j = (nt0 + tile) * 16 + nlocal;
            #pragma unroll
            for (int r = 0; r < 8; ++r) {
                const int row = r + crow;
                if (maskrow[row] != 0.0f) {
                    const int b = v0 + row;
                    float emit = label_score[((size_t)b * TT + t) * LL + j]
                               + penal_s[row * LL + j];
                    alpha[row * LL + j] = rowmax_s[row]
                                        + __logf(fmaxf(acc[r], 1e-30f)) + emit;
                }
            }
        }
        __syncthreads();
    }

    // ---- log_z[b] = logsumexp_j(alpha + end_t) ----
    {
        const int row = wid;
        float m = -3.4e38f;
        for (int q = lane; q < LL; q += 32)
            m = fmaxf(m, alpha[row * LL + q] + end_t[q]);
        for (int off = 16; off >= 1; off >>= 1)
            m = fmaxf(m, __shfl_xor(m, off));
        float s = 0.0f;
        for (int q = lane; q < LL; q += 32)
            s += __expf(alpha[row * LL + q] + end_t[q] - m);
        for (int off = 16; off >= 1; off >>= 1)
            s += __shfl_xor(s, off);
        if (lane == 0) logz_s[row] = m + __logf(s);
    }
    __syncthreads();

    // ---- gold path score per row (cheap serial gathers) ----
    if (tid < ROWS) {
        const int row = tid, b = v0 + row;
        const int* tg = &tags[b * TT];
        const int* mk = &mask[b * TT];
        float emit_sum = 0.0f, trans_sum = 0.0f;
        int len = 0, prev = tg[0];
        for (int t2 = 0; t2 < TT; ++t2) {
            const float mf = (mk[t2] != 0) ? 1.0f : 0.0f;
            len += (mk[t2] != 0) ? 1 : 0;
            const int g = tg[t2];
            emit_sum += mf * (label_score[((size_t)b * TT + t2) * LL + g]
                              + penal_s[row * LL + g]);
            if (t2 >= 1) trans_sum += mf * transitions[prev * LL + g];
            prev = g;
        }
        const int last = tg[(len > 0 ? len : 1) - 1];
        const float gold = start_t[tg[0]] + emit_sum + trans_sum + end_t[last];
        contrib[row] = logz_s[row] - gold;
    }
    __syncthreads();
    if (tid == 0) {
        float s = 0.0f;
        for (int r = 0; r < ROWS; ++r) s += contrib[r];
        partials[blockIdx.x] = s;
    }
}

// ---------------------------------------------------------------------------
// Kernel 4: reduce per-WG partials to the scalar output
// ---------------------------------------------------------------------------
__global__ void finalize_kernel(const float* __restrict__ partials, float* __restrict__ out)
{
    if (threadIdx.x == 0) {
        float s = 0.0f;
        for (int w = 0; w < NWG; ++w) s += partials[w];
        out[0] = s;
    }
}

// ---------------------------------------------------------------------------
extern "C" void kernel_launch(void* const* d_in, const int* in_sizes, int n_in,
                              void* d_out, int out_size, void* d_ws, size_t ws_size,
                              hipStream_t stream)
{
    const float* label_score = (const float*)d_in[0];
    const int*   tags        = (const int*)  d_in[1];
    const int*   mask        = (const int*)  d_in[2];
    const int*   semlink     = (const int*)  d_in[3];
    const int*   srl_b2i     = (const int*)  d_in[4];
    const int*   vn_b2i      = (const int*)  d_in[5];
    const int*   srl2c       = (const int*)  d_in[6];
    const int*   vn2c        = (const int*)  d_in[7];
    const int*   content     = (const int*)  d_in[8];
    const float* transitions = (const float*)d_in[9];
    const float* start_t     = (const float*)d_in[10];
    const float* end_t       = (const float*)d_in[11];
    float* out = (float*)d_out;

    // workspace layout
    char* ws = (char*)d_ws;
    float*     penal    = (float*)ws;                                     // 256 KB
    _Float16*  Bt       = (_Float16*)(ws + (size_t)NV * LL * 4);          // 512 KB
    float*     partials = (float*)(ws + (size_t)NV * LL * 4
                                      + (size_t)LL * LL * 2);             // 32 B

    semlink_kernel<<<NV, LL, 0, stream>>>(semlink, srl_b2i, vn_b2i,
                                          srl2c, vn2c, content, penal);
    expT_kernel<<<LL, LL, 0, stream>>>(transitions, Bt);
    crf_scan_kernel<<<NWG, NTHR, LDS_TOTAL, stream>>>(label_score, tags, mask,
                                                      transitions, start_t, end_t,
                                                      penal, Bt, partials);
    finalize_kernel<<<1, 32, 0, stream>>>(partials, out);
}
